// MLP_NW_dist_att_14370960572465
// MI455X (gfx1250) — compile-verified, matrix operand order checked by reading, not run on hardware
//
#include <hip/hip_runtime.h>
#include <hip/hip_bf16.h>

typedef __attribute__((ext_vector_type(16))) _Float16 v16h;
typedef __attribute__((ext_vector_type(8)))  float    v8f;

#define BB    4
#define NN    2048
#define MM    2048
#define INDIM 3
#define HH    256
#define DD    64
#define KV    32

#define TN 64
#define TM 64
#define NCHUNK (NN / TN)

#define VT_PITCH 68   // padded f16 pitch (avoid LDS bank conflicts on transpose writes)
#define PT_PITCH 68
#define S2_PITCH 65

// ---------------------------------------------------------------------------
// Phase 1: per-point MLP (3 -> 256 -> ReLU -> 64). Weights staged in LDS;
// each thread produces one point's 64-d embedding (64 f32 accumulators).
// ---------------------------------------------------------------------------
__global__ __launch_bounds__(64) void mlp_points_kernel(
    const float* __restrict__ coords,
    const float* __restrict__ W1, const float* __restrict__ b1,
    const float* __restrict__ W2, const float* __restrict__ b2,
    float* __restrict__ outF, int npts)
{
    __shared__ float W1s[INDIM * HH];
    __shared__ float b1s[HH];
    __shared__ float W2s[HH * DD];
    __shared__ float b2s[DD];

    const int tid = threadIdx.x;
    for (int i = tid; i < INDIM * HH; i += 64) W1s[i] = W1[i];
    for (int i = tid; i < HH;         i += 64) b1s[i] = b1[i];
    for (int i = tid; i < HH * DD;    i += 64) W2s[i] = W2[i];
    if (tid < DD) b2s[tid] = b2[tid];
    __syncthreads();

    const int  pt    = blockIdx.x * 64 + tid;
    const bool valid = (pt < npts);
    const int  src   = valid ? pt : 0;

    const float c0 = coords[(size_t)src * 3 + 0];
    const float c1 = coords[(size_t)src * 3 + 1];
    const float c2 = coords[(size_t)src * 3 + 2];

    float acc[DD];
#pragma unroll
    for (int d = 0; d < DD; ++d) acc[d] = b2s[d];

    for (int j = 0; j < HH; ++j) {
        float h = fmaf(c0, W1s[j], fmaf(c1, W1s[HH + j], fmaf(c2, W1s[2 * HH + j], b1s[j])));
        h = fmaxf(h, 0.0f);           // ReLU
#pragma unroll
        for (int d = 0; d < DD; ++d) acc[d] = fmaf(h, W2s[j * DD + d], acc[d]);
    }

    if (valid) {
#pragma unroll
        for (int d = 0; d < DD; ++d) outF[(size_t)pt * DD + d] = acc[d];
    }
}

// ---------------------------------------------------------------------------
// Phase 2: fused L1-distance attention.
// Block = (batch b, 64-query tile). 256 threads = 8 waves.
// Each wave owns one 16x16 (m,k) f32 accumulator tile -> covers 64m x 32k.
// Online softmax over the key axis; p^T @ V via v_wmma_f32_16x16x32_f16.
// ---------------------------------------------------------------------------
__global__ __launch_bounds__(256) void fused_l1_attn_kernel(
    const float* __restrict__ Kd,   // [B,N,D] f32 keys
    const float* __restrict__ Qd,   // [B,M,D] f32 queries
    const float* __restrict__ Vf,   // [B,N,KV] f32 values
    float* __restrict__ Out)        // [B,M,KV] f32
{
    __shared__ float    Ks[TN * DD];           // 16 KB key chunk
    __shared__ _Float16 Vt[KV * VT_PITCH];     // V chunk transposed: Vt[k][n]
    __shared__ _Float16 Pt[TM * PT_PITCH];     // softmax probs transposed: Pt[m][n]
    __shared__ float    S2[TN * S2_PITCH];     // -0.5*s^2 scratch [n][m]
    __shared__ float    mxs[TM], ls[TM], corrs[TM];

    const int tid  = threadIdx.x;
    const int lane = tid & 31;
    const int wave = tid >> 5;       // 0..7
    const int mw   = wave >> 1;      // 0..3 : 16-row m subtile
    const int kw   = wave & 1;       // 0..1 : 16-col k subtile
    const int b    = blockIdx.y;
    const int m0   = blockIdx.x * TM;

    // Query row for the distance core: thread (mi) holds Q[m0+mi][:] in regs.
    const int mi = tid & 63;
    const int ng = tid >> 6;         // 0..3 : 16-key group
    float q[DD];
    {
        const float* qp = Qd + ((size_t)b * MM + (m0 + mi)) * DD;
#pragma unroll
        for (int d = 0; d < DD; ++d) q[d] = qp[d];
    }

    if (tid < TM) { mxs[tid] = -1e30f; ls[tid] = 0.0f; }

    v8f c = {};   // f32 WMMA accumulator (16x16)

    const int hi8 = (lane >= 16) ? 8 : 0;   // high-lane K/M offset per WMMA layout

    for (int nc = 0; nc < NCHUNK; ++nc) {
        __syncthreads();   // previous iteration's Pt/Vt fully consumed

        // ---- stage K chunk (contiguous 64x64 f32) via float4 ----
        {
            const float4* src = (const float4*)(Kd + ((size_t)b * NN + (size_t)nc * TN) * DD);
            float4*       dst = (float4*)Ks;
#pragma unroll
            for (int t = 0; t < (TN * DD / 4) / 256; ++t)
                dst[tid + t * 256] = src[tid + t * 256];
            if (nc + 1 < NCHUNK) {
                const float* nxt = Kd + ((size_t)b * NN + (size_t)(nc + 1) * TN) * DD;
                __builtin_prefetch(nxt + tid * 4, 0, 1);   // global_prefetch_b8
            }
        }
        // ---- stage V chunk transposed to f16: Vt[k][n] ----
        {
#pragma unroll
            for (int t = 0; t < (TN * KV) / 256; ++t) {
                int idx = tid + t * 256;
                int n   = idx >> 5;    // 0..63
                int k   = idx & 31;    // coalesced in k
                Vt[k * VT_PITCH + n] =
                    (_Float16)Vf[((size_t)b * NN + (size_t)nc * TN + n) * KV + k];
            }
        }
        __syncthreads();

        // ---- L1 distance core: thread handles 16 keys for its query mi ----
#pragma unroll 1
        for (int i = 0; i < 16; ++i) {
            const int    n  = ng * 16 + i;
            const float* kr = &Ks[n * DD];
            float s = 0.0f;
#pragma unroll
            for (int d = 0; d < DD; ++d) s += fabsf(kr[d] - q[d]);
            S2[n * S2_PITCH + mi] = -0.5f * s * s;
        }
        __syncthreads();

        // ---- online softmax over key axis (one thread per query) ----
        if (tid < TM) {
            const int m = tid;
            float cm = -1e30f;
#pragma unroll
            for (int n = 0; n < TN; ++n) cm = fmaxf(cm, S2[n * S2_PITCH + m]);
            const float old  = mxs[m];
            const float nm   = fmaxf(old, cm);
            const float corr = __expf(old - nm);
            float sum = 0.0f;
#pragma unroll
            for (int n = 0; n < TN; ++n) {
                const float p = __expf(S2[n * S2_PITCH + m] - nm);
                Pt[m * PT_PITCH + n] = (_Float16)p;
                sum += p;
            }
            ls[m]    = ls[m] * corr + sum;
            mxs[m]   = nm;
            corrs[m] = corr;
        }
        __syncthreads();

        // ---- rescale accumulator, then WMMA-accumulate p^T(16x64) @ V(64x16) ----
        {
#pragma unroll
            for (int r = 0; r < 8; ++r) c[r] *= corrs[mw * 16 + r + hi8];

            const _Float16* pr = &Pt[(mw * 16 + (lane & 15)) * PT_PITCH];
            const _Float16* vr = &Vt[(kw * 16 + (lane & 15)) * VT_PITCH];
#pragma unroll
            for (int ks = 0; ks < 2; ++ks) {
                const int nb = ks * 32;
                v16h a, bmat;
                // 16-bit A/B fragment layout: lanes 0-15 hold K=0..7 / 16..23,
                // lanes 16-31 hold K=8..15 / 24..31 (VGPR0..3 then VGPR4..7).
#pragma unroll
                for (int j = 0; j < 8; ++j) {
                    a[j]        = pr[nb + hi8 + j];
                    a[8 + j]    = pr[nb + 16 + hi8 + j];
                    bmat[j]     = vr[nb + hi8 + j];
                    bmat[8 + j] = vr[nb + 16 + hi8 + j];
                }
                c = __builtin_amdgcn_wmma_f32_16x16x32_f16(
                        false, a, false, bmat, (short)0, c, false, false);
            }
        }
    }

    // ---- normalize by softmax denominator and write out ----
    {
        const int k = kw * 16 + (lane & 15);
#pragma unroll
        for (int r = 0; r < 8; ++r) {
            const int   ml    = mw * 16 + r + hi8;   // C layout: M = r + 8*(lane>=16)
            const float denom = ls[ml];
            Out[((size_t)b * MM + (m0 + ml)) * KV + k] = c[r] / denom;
        }
    }
}

// ---------------------------------------------------------------------------
extern "C" void kernel_launch(void* const* d_in, const int* in_sizes, int n_in,
                              void* d_out, int out_size, void* d_ws, size_t ws_size,
                              hipStream_t stream) {
    (void)in_sizes; (void)n_in; (void)out_size; (void)ws_size;

    const float* coords_f = (const float*)d_in[0];   // [B,N,3]
    const float* values_f = (const float*)d_in[1];   // [B,N,32]
    const float* coords_t = (const float*)d_in[2];   // [B,M,3]
    const float* Wk1 = (const float*)d_in[3];
    const float* bk1 = (const float*)d_in[4];
    const float* Wk2 = (const float*)d_in[5];
    const float* bk2 = (const float*)d_in[6];
    const float* Wq1 = (const float*)d_in[7];
    const float* bq1 = (const float*)d_in[8];
    const float* Wq2 = (const float*)d_in[9];
    const float* bq2 = (const float*)d_in[10];

    float* out = (float*)d_out;                      // [B,M,32]
    float* Kd  = (float*)d_ws;                       // [B,N,64] f32 keys
    float* Qd  = Kd + (size_t)BB * NN * DD;          // [B,M,64] f32 queries

    // Phase 1: key / query embeddings
    mlp_points_kernel<<<dim3((BB * NN + 63) / 64), 64, 0, stream>>>(
        coords_f, Wk1, bk1, Wk2, bk2, Kd, BB * NN);
    mlp_points_kernel<<<dim3((BB * MM + 63) / 64), 64, 0, stream>>>(
        coords_t, Wq1, bq1, Wq2, bq2, Qd, BB * MM);

    // Phase 2: fused distance-softmax-weighted-sum
    fused_l1_attn_kernel<<<dim3(MM / TM, BB), 256, 0, stream>>>(
        Kd, Qd, values_f, out);
}
